// cls_model_14508399526635
// MI455X (gfx1250) — matrix operand order repaired and checked by reading, not compiled, not run on hardware
//
#include <hip/hip_runtime.h>

// ---------- types ----------
typedef __bf16 bf16_t;
typedef __attribute__((ext_vector_type(16))) __bf16 v16bf;
typedef __attribute__((ext_vector_type(8)))  __bf16 v8bf;
typedef __attribute__((ext_vector_type(8)))  float  v8f;
typedef int v4i32 __attribute__((vector_size(16)));

#define AS1 __attribute__((address_space(1)))
#define AS3 __attribute__((address_space(3)))

union BF16V { v16bf v; v8bf h[2]; };

__device__ __forceinline__ bf16_t f2bf(float fv) {
    union { float f; unsigned u; } in; in.f = fv;
    unsigned u = in.u;
    unsigned r = u + 0x7FFFu + ((u >> 16) & 1u);   // round-to-nearest-even
    union { unsigned short s; bf16_t b; } out;
    out.s = (unsigned short)(r >> 16);
    return out.b;
}

// ---------- async global->LDS (CDNA5 ASYNCcnt path) ----------
#ifndef __has_builtin
#define __has_builtin(x) 0
#endif
#if __has_builtin(__builtin_amdgcn_global_load_async_to_lds_b128)
#define HAVE_ASYNC_LDS 1
#else
#define HAVE_ASYNC_LDS 0
#endif

__device__ __forceinline__ void async_cp16(const bf16_t* g, bf16_t* l) {
#if HAVE_ASYNC_LDS
    __builtin_amdgcn_global_load_async_to_lds_b128(
        (AS1 v4i32*)g, (AS3 v4i32*)l, 0, 0);
#else
    *(uint4*)l = *(const uint4*)g;   // synchronous fallback
#endif
}

__device__ __forceinline__ void wait_async_lds() {
#if HAVE_ASYNC_LDS
#if __has_builtin(__builtin_amdgcn_s_wait_asynccnt)
    __builtin_amdgcn_s_wait_asynccnt(0);
#else
    asm volatile("s_wait_asynccnt 0x0" ::: "memory");
#endif
#endif
}

// ---------- problem constants ----------
#define BATCH 4
#define NPTS  16384
#define NC    1024
#define KNN   32
#define LCOLS (NC * KNN)        // 32768 columns per batch
#define SLAB  128               // columns per workgroup in the conv kernel
#define S64   72                // padded LDS stride for 64-ch tensors (bf16 elems)
#define S128  136               // padded LDS stride for 128-ch tensors

// ---------- workspace layout (bytes) ----------
#define WS_CENT 0                                   // B*1024*3 f32 = 49152
#define WS_ACT0 49152                               // B*32768*3 f32 = 1572864
#define WS_G    (49152 + 1572864)                   // B*1024 int/f32 = 16384
#define WS_W5BF (WS_G + 16384)                      // 1024*136 bf16 = 278528

// =====================================================================
// K0: zero max-pool accumulator + pre-convert W5 to padded bf16 image
// (rows padded to stride 136 so one 256-row chunk is contiguous 69632B)
// =====================================================================
__global__ void k0_prep(const float* __restrict__ W5, bf16_t* __restrict__ w5bf,
                        int* __restrict__ g) {
    const int i = blockIdx.x * blockDim.x + threadIdx.x;   // 131072 threads
    if (i < BATCH * NC) g[i] = 0;
    const int r = i >> 7, k = i & 127;
    w5bf[r * S128 + k] = f2bf(W5[i]);
}

// =====================================================================
// K1: FPS. One workgroup per batch, x resident in LDS, 1024 sequential
// argmax iterations, shfl + cross-wave tree reduction (2 barriers/step).
// =====================================================================
#define K1_SMEM (NPTS*3*4 + 32*4 + 32*4 + 16)
__global__ void k1_fps(const float* __restrict__ x, const int* __restrict__ far0,
                       float* __restrict__ cent) {
    extern __shared__ char smem[];
    float* xl   = (float*)smem;                       // 16384*3
    float* redd = (float*)(smem + NPTS*3*4);          // 32
    int*   redi = (int*)  (smem + NPTS*3*4 + 128);    // 32
    int*   win  = (int*)  (smem + NPTS*3*4 + 256);

    const int tid = threadIdx.x, b = blockIdx.x;
    const int lane = tid & 31, wv = tid >> 5;

    for (int i = tid; i < NPTS*3; i += 1024) xl[i] = x[b*NPTS*3 + i];
    __syncthreads();

    int far = far0[b];
    for (int t = 0; t < NC; ++t) {
        const float cx = xl[far*3+0], cy = xl[far*3+1], cz = xl[far*3+2];
        if (tid == 0) {
            cent[(b*NC+t)*3+0] = cx; cent[(b*NC+t)*3+1] = cy; cent[(b*NC+t)*3+2] = cz;
        }
        float bd = -1.0f; int bi = 0;
        for (int p = tid; p < NPTS; p += 1024) {
            float dx = xl[p*3+0]-cx, dy = xl[p*3+1]-cy, dz = xl[p*3+2]-cz;
            float d = dx*dx + dy*dy + dz*dz;
            if (d > bd) { bd = d; bi = p; }
        }
        for (int m = 16; m >= 1; m >>= 1) {
            float od = __shfl_xor(bd, m, 32);
            int   oi = __shfl_xor(bi, m, 32);
            if (od > bd || (od == bd && oi < bi)) { bd = od; bi = oi; }
        }
        if (lane == 0) { redd[wv] = bd; redi[wv] = bi; }
        __syncthreads();
        if (tid < 32) {
            float d2 = redd[tid]; int i2 = redi[tid];
            for (int m = 16; m >= 1; m >>= 1) {
                float od = __shfl_xor(d2, m, 32);
                int   oi = __shfl_xor(i2, m, 32);
                if (od > d2 || (od == d2 && oi < i2)) { d2 = od; i2 = oi; }
            }
            if (tid == 0) *win = i2;
        }
        __syncthreads();
        far = *win;
    }
}

// =====================================================================
// K2: KNN + grouping. WG handles 16 centroids of one batch; x + one
// distance row in LDS; top-32 via masked argmin passes; writes grouped
// points act0[b*32768 + cent*32 + k][3].
// =====================================================================
#define K2_SMEM (NPTS*3*4 + NPTS*4 + 32 + 32 + 16 + KNN*4)
__global__ void k2_knn(const float* __restrict__ x, const float* __restrict__ cent,
                       float* __restrict__ act0) {
    extern __shared__ char smem[];
    float* xl   = (float*)smem;                           // 49152 f
    float* d2   = (float*)(smem + NPTS*3*4);              // 16384 f
    float* redd = (float*)(smem + NPTS*3*4 + NPTS*4);     // 8
    int*   redi = (int*)  (smem + NPTS*3*4 + NPTS*4 + 32);// 8
    int*   idxl = (int*)  (smem + NPTS*3*4 + NPTS*4 + 80);// 32

    const int tid = threadIdx.x;
    const int b  = blockIdx.x >> 6;
    const int cg = blockIdx.x & 63;
    const int lane = tid & 31, wv = tid >> 5;

    for (int i = tid; i < NPTS*3; i += 256) xl[i] = x[b*NPTS*3 + i];
    __syncthreads();

    for (int ci = 0; ci < 16; ++ci) {
        const int cidx = cg*16 + ci;
        const float cx = cent[(b*NC+cidx)*3+0];
        const float cy = cent[(b*NC+cidx)*3+1];
        const float cz = cent[(b*NC+cidx)*3+2];
        for (int p = tid; p < NPTS; p += 256) {
            float dx = xl[p*3+0]-cx, dy = xl[p*3+1]-cy, dz = xl[p*3+2]-cz;
            d2[p] = dx*dx + dy*dy + dz*dz;
        }
        __syncthreads();
        for (int kk = 0; kk < KNN; ++kk) {
            float bd = 3.0e38f; int bi = 0;
            for (int p = tid; p < NPTS; p += 256) {
                float d = d2[p];
                if (d < bd) { bd = d; bi = p; }
            }
            for (int m = 16; m >= 1; m >>= 1) {
                float od = __shfl_xor(bd, m, 32);
                int   oi = __shfl_xor(bi, m, 32);
                if (od < bd || (od == bd && oi < bi)) { bd = od; bi = oi; }
            }
            if (lane == 0) { redd[wv] = bd; redi[wv] = bi; }
            __syncthreads();
            if (tid == 0) {
                float gb = redd[0]; int gi = redi[0];
                for (int wj = 1; wj < 8; ++wj)
                    if (redd[wj] < gb || (redd[wj] == gb && redi[wj] < gi)) { gb = redd[wj]; gi = redi[wj]; }
                idxl[kk] = gi;
                d2[gi] = 3.0e38f;
            }
            __syncthreads();
        }
        if (tid < KNN) {
            const int gi = idxl[tid];
            const int colg = b*LCOLS + cidx*KNN + tid;
            act0[colg*3+0] = xl[gi*3+0];
            act0[colg*3+1] = xl[gi*3+1];
            act0[colg*3+2] = xl[gi*3+2];
        }
        __syncthreads();
    }
}

// =====================================================================
// K3: fused conv stack (3->64->64->64->128->1024) + global max-pool.
// 256 threads (8 waves), 128 columns per WG (16 cols per wave).
// bf16 WMMA 16x16x32, f32 accumulate. W5 streamed through LDS with
// async double-buffering (GLOBAL_LOAD_ASYNC_TO_LDS_B128 / ASYNCcnt).
// =====================================================================
#define OFF_W2   0          //  64*72*2  = 9216
#define OFF_W3   9216       //  64*72*2  = 9216
#define OFF_W4   18432      // 128*72*2  = 18432
#define OFF_W5A  36864      // 256*136*2 = 69632  (chunk buffer 0)
#define W5CHUNK  69632      // bytes per chunk buffer
#define OFF_AA   176128     // 128*72*2
#define OFF_AB   194560     // 128*72*2
#define OFF_AC   212992     // 128*136*2
#define OFF_BIAS 247808     // (64+64+128+1024)*4
#define OFF_GP   252928     // 1024*4
#define K3_SMEM  257024

template<int CIN, int COUT, int SIN, int SOUT, int SW>
__device__ __forceinline__ void conv_layer(const bf16_t* __restrict__ W,
                                           const float* __restrict__ bias,
                                           const bf16_t* actIn, bf16_t* actOut,
                                           int lane) {
    const int l = lane & 15, half = lane >> 4;
    constexpr int NK = CIN / 32;
    // B fragments (this lane's column) are invariant over the M loop: hoist.
    BF16V bb[NK];
    #pragma unroll
    for (int kt = 0; kt < NK; ++kt) {
        const bf16_t* br = actIn + l * SIN + kt*32 + half*8;
        bb[kt].h[0] = *(const v8bf*)(br);
        bb[kt].h[1] = *(const v8bf*)(br + 16);
    }
    for (int m0 = 0; m0 < COUT; m0 += 16) {
        v8f c;
        #pragma unroll
        for (int v = 0; v < 8; ++v) c[v] = bias[m0 + half*8 + v];
        #pragma unroll
        for (int kt = 0; kt < NK; ++kt) {
            // A fragment: row m0+l; K in {8h..8h+7} U {16+8h..16+8h+7}
            const bf16_t* ar = W + (m0 + l) * SW + kt*32;
            BF16V a;
            a.h[0] = *(const v8bf*)(ar + half*8);
            a.h[1] = *(const v8bf*)(ar + 16 + half*8);
            c = __builtin_amdgcn_wmma_f32_16x16x32_bf16(false, a.v, false, bb[kt].v,
                                                        (short)0, c, false, false);
        }
        v8bf o;
        #pragma unroll
        for (int v = 0; v < 8; ++v) o[v] = f2bf(fmaxf(c[v], 0.0f));
        *(v8bf*)(actOut + l * SOUT + m0 + half*8) = o;   // rows m0+8h..+7, col l
    }
}

__global__ void k3_conv(const float* __restrict__ act0,
                        const float* __restrict__ W1g, const float* __restrict__ b1g,
                        const float* __restrict__ W2g, const float* __restrict__ b2g,
                        const float* __restrict__ W3g, const float* __restrict__ b3g,
                        const float* __restrict__ W4g, const float* __restrict__ b4g,
                        const bf16_t* __restrict__ w5bf, const float* __restrict__ b5g,
                        int* __restrict__ g_i) {
    extern __shared__ char smem[];
    bf16_t* w2    = (bf16_t*)(smem + OFF_W2);
    bf16_t* w3    = (bf16_t*)(smem + OFF_W3);
    bf16_t* w4    = (bf16_t*)(smem + OFF_W4);
    bf16_t* actA  = (bf16_t*)(smem + OFF_AA);
    bf16_t* actB  = (bf16_t*)(smem + OFF_AB);
    bf16_t* actC  = (bf16_t*)(smem + OFF_AC);
    float*  biasL = (float*)(smem + OFF_BIAS);
    int*    gpart = (int*)  (smem + OFF_GP);

    const int tid  = threadIdx.x;
    const int b    = blockIdx.x >> 8;     // 256 slabs per batch
    const int slab = blockIdx.x & 255;

    // ---- kick off async stream of W5 chunk 0 (overlaps with all staging) ----
    {
        bf16_t* ldst = (bf16_t*)(smem + OFF_W5A);
        for (int j = 0; j < 17; ++j) {                   // 17*256 = 4352 x 16B
            const int idx = tid + j*256;
            async_cp16(w5bf + idx*8, ldst + idx*8);
        }
    }

    // ---- stage small weights/biases (f32 -> bf16), zero partial max ----
    for (int i = tid; i < 64*64;  i += 256) w2[(i>>6)*S64 + (i&63)] = f2bf(W2g[i]);
    for (int i = tid; i < 64*64;  i += 256) w3[(i>>6)*S64 + (i&63)] = f2bf(W3g[i]);
    for (int i = tid; i < 128*64; i += 256) w4[(i>>6)*S64 + (i&63)] = f2bf(W4g[i]);
    for (int i = tid; i < 64;   i += 256) biasL[i]       = b2g[i];
    for (int i = tid; i < 64;   i += 256) biasL[64+i]    = b3g[i];
    for (int i = tid; i < 128;  i += 256) biasL[128+i]   = b4g[i];
    for (int i = tid; i < 1024; i += 256) biasL[256+i]   = b5g[i];
    for (int i = tid; i < 1024; i += 256) gpart[i] = 0;
    __syncthreads();

    const int w = tid >> 5, lane = tid & 31, l = lane & 15, half = lane >> 4;

    // ---- layer 1 (3 -> 64): K too small for WMMA, do VALU ----
    {
        const int cglob = b*LCOLS + slab*SLAB + w*16 + l;
        const float x0 = act0[cglob*3+0], x1 = act0[cglob*3+1], x2 = act0[cglob*3+2];
        bf16_t* outc = actA + (w*16 + l) * S64;
        const int ch0 = half * 32;
        for (int j = 0; j < 32; ++j) {
            const int ch = ch0 + j;
            float h = W1g[ch*3+0]*x0 + W1g[ch*3+1]*x1 + W1g[ch*3+2]*x2 + b1g[ch];
            outc[ch] = f2bf(fmaxf(h, 0.0f));
        }
    }

    // per-wave activation bases (each wave owns its 16 columns -> no barriers)
    bf16_t* A  = actA + w*16*S64;
    bf16_t* Bb = actB + w*16*S64;
    bf16_t* Cc = actC + w*16*S128;

    conv_layer<64, 64,  S64, S64,  S64>(w2, biasL+0,   A,  Bb, lane);  // h1->h2
    conv_layer<64, 64,  S64, S64,  S64>(w3, biasL+64,  Bb, A,  lane);  // h2->h3
    conv_layer<64, 128, S64, S128, S64>(w4, biasL+128, A,  Cc, lane);  // h3->h4

    // ---- layer 5 (128 -> 1024): async double-buffered W5 chunks,
    //      B fragments (h4 column) hoisted for the entire layer ----
    BF16V bfr[4];
    #pragma unroll
    for (int kt = 0; kt < 4; ++kt) {
        const bf16_t* br = Cc + l * S128 + kt*32 + half*8;
        bfr[kt].h[0] = *(const v8bf*)(br);
        bfr[kt].h[1] = *(const v8bf*)(br + 16);
    }

    for (int cc = 0; cc < 4; ++cc) {
        const int cb = cc * 256;
        wait_async_lds();            // own async loads for chunk cc complete
        __syncthreads();             // all waves' loads visible
        if (cc < 3) {                // stream next chunk while computing
            const bf16_t* gsrc = w5bf + (cc + 1) * 256 * S128;
            bf16_t* ldst = (bf16_t*)(smem + OFF_W5A + ((cc + 1) & 1) * W5CHUNK);
            for (int j = 0; j < 17; ++j) {
                const int idx = tid + j*256;
                async_cp16(gsrc + idx*8, ldst + idx*8);
            }
        }
        const bf16_t* wb = (const bf16_t*)(smem + OFF_W5A + (cc & 1) * W5CHUNK);
        for (int m0 = 0; m0 < 256; m0 += 32) {      // dual accumulators
            v8f c0, c1;
            #pragma unroll
            for (int v = 0; v < 8; ++v) {
                c0[v] = biasL[256 + cb + m0      + half*8 + v];
                c1[v] = biasL[256 + cb + m0 + 16 + half*8 + v];
            }
            #pragma unroll
            for (int kt = 0; kt < 4; ++kt) {
                const bf16_t* ar0 = wb + (m0      + l) * S128 + kt*32;
                const bf16_t* ar1 = wb + (m0 + 16 + l) * S128 + kt*32;
                BF16V a0, a1;
                a0.h[0] = *(const v8bf*)(ar0 + half*8);
                a0.h[1] = *(const v8bf*)(ar0 + 16 + half*8);
                a1.h[0] = *(const v8bf*)(ar1 + half*8);
                a1.h[1] = *(const v8bf*)(ar1 + 16 + half*8);
                c0 = __builtin_amdgcn_wmma_f32_16x16x32_bf16(false, a0.v, false, bfr[kt].v,
                                                             (short)0, c0, false, false);
                c1 = __builtin_amdgcn_wmma_f32_16x16x32_bf16(false, a1.v, false, bfr[kt].v,
                                                             (short)0, c1, false, false);
            }
            // max over the 16 columns of this wave-half, then LDS atomic max.
            // relu folded into max (gpart init 0; int-bit max correct for >=0).
            #pragma unroll
            for (int v = 0; v < 8; ++v) {
                float x0 = c0[v], x1 = c1[v];
                x0 = fmaxf(x0, __shfl_xor(x0, 1, 32));
                x1 = fmaxf(x1, __shfl_xor(x1, 1, 32));
                x0 = fmaxf(x0, __shfl_xor(x0, 2, 32));
                x1 = fmaxf(x1, __shfl_xor(x1, 2, 32));
                x0 = fmaxf(x0, __shfl_xor(x0, 4, 32));
                x1 = fmaxf(x1, __shfl_xor(x1, 4, 32));
                x0 = fmaxf(x0, __shfl_xor(x0, 8, 32));
                x1 = fmaxf(x1, __shfl_xor(x1, 8, 32));
                if (l == 0) {
                    atomicMax(&gpart[cb + m0      + half*8 + v], __float_as_int(x0));
                    atomicMax(&gpart[cb + m0 + 16 + half*8 + v], __float_as_int(x1));
                }
            }
        }
    }
    __syncthreads();
    for (int i = tid; i < NC; i += 256)
        atomicMax(&g_i[b*NC + i], gpart[i]);
}

// =====================================================================
// K4: FC head 1024 -> 512 -> 256 -> 3 (B=4, trivial)
// =====================================================================
__global__ void k4_head(const float* __restrict__ g,
                        const float* __restrict__ fw1, const float* __restrict__ fb1,
                        const float* __restrict__ fw2, const float* __restrict__ fb2,
                        const float* __restrict__ fw3, const float* __restrict__ fb3,
                        float* __restrict__ out) {
    __shared__ float gin[1024];
    __shared__ float g1[512];
    __shared__ float g2[256];
    const int b = blockIdx.x, tid = threadIdx.x;
    for (int i = tid; i < 1024; i += 512) gin[i] = g[b*NC + i];
    __syncthreads();
    if (tid < 512) {
        float a = fb1[tid];
        const float* wr = fw1 + tid*1024;
        for (int i = 0; i < 1024; ++i) a += wr[i] * gin[i];
        g1[tid] = fmaxf(a, 0.0f);
    }
    __syncthreads();
    if (tid < 256) {
        float a = fb2[tid];
        const float* wr = fw2 + tid*512;
        for (int i = 0; i < 512; ++i) a += wr[i] * g1[i];
        g2[tid] = fmaxf(a, 0.0f);
    }
    __syncthreads();
    if (tid < 3) {
        float a = fb3[tid];
        const float* wr = fw3 + tid*256;
        for (int i = 0; i < 256; ++i) a += wr[i] * g2[i];
        out[b*3 + tid] = a;
    }
}

// =====================================================================
extern "C" void kernel_launch(void* const* d_in, const int* in_sizes, int n_in,
                              void* d_out, int out_size, void* d_ws, size_t ws_size,
                              hipStream_t stream) {
    (void)in_sizes; (void)n_in; (void)out_size; (void)ws_size;
    const float* x    = (const float*)d_in[0];
    const int*   far0 = (const int*)  d_in[1];
    const float* W1 = (const float*)d_in[2];  const float* b1 = (const float*)d_in[3];
    const float* W2 = (const float*)d_in[4];  const float* b2 = (const float*)d_in[5];
    const float* W3 = (const float*)d_in[6];  const float* b3 = (const float*)d_in[7];
    const float* W4 = (const float*)d_in[8];  const float* b4 = (const float*)d_in[9];
    const float* W5 = (const float*)d_in[10]; const float* b5 = (const float*)d_in[11];
    const float* fw1 = (const float*)d_in[12]; const float* fb1 = (const float*)d_in[13];
    const float* fw2 = (const float*)d_in[14]; const float* fb2 = (const float*)d_in[15];
    const float* fw3 = (const float*)d_in[16]; const float* fb3 = (const float*)d_in[17];

    char* ws = (char*)d_ws;
    float*  cent = (float*)(ws + WS_CENT);
    float*  act0 = (float*)(ws + WS_ACT0);
    int*    g_i  = (int*)  (ws + WS_G);
    bf16_t* w5bf = (bf16_t*)(ws + WS_W5BF);

    k0_prep<<<512, 256, 0, stream>>>(W5, w5bf, g_i);                 // 131072 thr
    k1_fps <<<BATCH, 1024, K1_SMEM, stream>>>(x, far0, cent);
    k2_knn <<<BATCH*64, 256, K2_SMEM, stream>>>(x, cent, act0);
    k3_conv<<<BATCH*256, 256, K3_SMEM, stream>>>(act0, W1, b1, W2, b2, W3, b3,
                                                 W4, b4, w5bf, b5, g_i);
    k4_head<<<BATCH, 512, 0, stream>>>((const float*)g_i, fw1, fb1, fw2, fb2,
                                       fw3, fb3, (float*)d_out);
}
